// PatchMoEBlock_824633721567
// MI455X (gfx1250) — compile-verified, matrix operand order checked by reading, not run on hardware
//
#include <hip/hip_runtime.h>
#include <hip/hip_bf16.h>
#include <math.h>

// ---------------- problem constants ----------------
#define BB   4
#define SS   16384        // 128*128
#define DD   192
#define EE   8
#define TT   1024         // 32*32 patches per image
#define PP   16           // 4*4 tokens per patch
#define II   384          // 2*D
#define KSEL 256          // top-k per expert = T*CAP/E
#define IMGW 128
#define TW_  32           // patches per row
#define GTOK 4            // patches per FFN block (M = 64)

typedef _Float16 v16h __attribute__((ext_vector_type(16)));
typedef _Float16 v8h  __attribute__((ext_vector_type(8)));
typedef float    v8f  __attribute__((ext_vector_type(8)));
typedef float    v4f  __attribute__((ext_vector_type(4)));

__device__ __forceinline__ v16h concat8(v8h lo, v8h hi) {
  return __builtin_shufflevector(lo, hi, 0,1,2,3,4,5,6,7,8,9,10,11,12,13,14,15);
}

// load a 16-half fragment (two b128) from a per-lane base + compile-time-friendly offset (halves)
__device__ __forceinline__ v16h load_frag(const _Float16* __restrict__ p) {
  v8h lo = *(const v8h*)p;
  v8h hi = *(const v8h*)(p + 8);
  return concat8(lo, hi);
}
// A-operand: 8 contiguous + 8 contiguous at +16 halves
__device__ __forceinline__ v16h load_afrag(const _Float16* __restrict__ p) {
  v8h lo = *(const v8h*)p;
  v8h hi = *(const v8h*)(p + 16);
  return concat8(lo, hi);
}

__device__ __forceinline__ v8f wmma16(v16h a, v16h b, v8f c) {
  // (neg_a, A, neg_b, B, c_mod, C, reuse_a, reuse_b)
  return __builtin_amdgcn_wmma_f32_16x16x32_f16(false, a, false, b, (short)0, c, false, false);
}

// ---------------- init: selection slots = -1, output = 0 ----------------
__global__ void init_selv_kernel(float* __restrict__ selv) {
  int i = blockIdx.x * 256 + threadIdx.x;
  if (i < BB * TT * EE) selv[i] = -1.0f;
}

__global__ void zero_out_kernel(v4f* __restrict__ out) {
  int i = blockIdx.x * 256 + threadIdx.x;   // BB*SS*DD/4 = 3,145,728 vec4
  v4f z; z[0]=0.f; z[1]=0.f; z[2]=0.f; z[3]=0.f;
  out[i] = z;
}

// ---------------- weight f16 pack into B-fragment layout ----------------
// src: [E][N][K] row-major f32.  dst id = (((e*KS+ks)*NT+nt)*32+lane)*16+h
// B layout (mirror of A): lane<16 col=lane K{0..7,16..23}; lane>=16 col=lane-16 K{8..15,24..31}
__global__ void prep_w_kernel(const float* __restrict__ src, _Float16* __restrict__ dst,
                              int KS, int NT, int N, int K, int total) {
  int id = blockIdx.x * 256 + threadIdx.x;
  if (id >= total) return;
  int h    = id & 15;
  int lane = (id >> 4) & 31;
  int rest = id >> 9;           // e*KS*NT + ks*NT + nt
  int nt = rest % NT; rest /= NT;
  int ks = rest % KS;
  int e  = rest / KS;
  int col = lane & 15;
  int kk  = (h & 7) + ((h >> 3) << 4) + ((lane >> 4) << 3);
  int k = ks * 32 + kk;
  int n = nt * 16 + col;
  dst[id] = (_Float16)src[((size_t)e * N + n) * K + k];
}

// ---------------- fold complexity path:  Weff = Wg2 @ Wc,  beff = Wg2·bc + bg ----------------
__global__ __launch_bounds__(DD) void prep_weff_kernel(
    const float* __restrict__ Wc, const float* __restrict__ bc,
    const float* __restrict__ Wg, const float* __restrict__ bg,
    float* __restrict__ Weff, float* __restrict__ beff) {
  int e  = blockIdx.x;          // E blocks x D threads
  int dd = threadIdx.x;
  float s = 0.f;
  for (int d = 0; d < DD; ++d)
    s = fmaf(Wg[e * (2 * DD) + DD + d], Wc[d * DD + dd], s);   // coalesced in dd
  Weff[e * DD + dd] = s;
  if (dd == 0) {
    float bsum = bg[e];
    for (int d = 0; d < DD; ++d) bsum = fmaf(Wg[e * (2 * DD) + DD + d], bc[d], bsum);
    beff[e] = bsum;
  }
}

// ---------------- patch stats + LN + gate softmax (complexity folded into Weff) ----------------
__global__ __launch_bounds__(192) void stats_gate_kernel(
    const float* __restrict__ x,  const float* __restrict__ ln_g, const float* __restrict__ ln_b,
    const float* __restrict__ Wg, const float* __restrict__ Weff, const float* __restrict__ beff,
    _Float16* __restrict__ xp, float* __restrict__ aff)
{
  int bt = blockIdx.x;                 // b*T + t
  int b  = bt >> 10, t = bt & (TT - 1);
  int th_ = t >> 5, tw_ = t & (TW_ - 1);
  int d = threadIdx.x;

  __shared__ float smean[DD];
  __shared__ float spart[EE][DD];
  __shared__ float sred[2];
  __shared__ float slog[EE];

  const float* xb  = x + (size_t)b * SS * DD;
  _Float16*   xpb  = xp + (size_t)bt * PP * DD;

  float v[PP];
  float sum = 0.f;
#pragma unroll
  for (int p = 0; p < PP; ++p) {
    int row = th_ * 4 + (p >> 2), col = tw_ * 4 + (p & 3);
    float val = xb[((size_t)(row * IMGW + col)) * DD + d];
    v[p] = val; sum += val;
    xpb[p * DD + d] = (_Float16)val;
  }
  float pm = sum * (1.f / 16.f);
  float sq = 0.f;
#pragma unroll
  for (int p = 0; p < PP; ++p) { float z = v[p] - pm; sq += z * z; }
  float pv = sq * (1.f / 16.f);
  smean[d] = pm;
  __syncthreads();

  if (d == 0) {
    float mu = 0.f;
    for (int i = 0; i < DD; ++i) mu += smean[i];
    mu *= (1.f / DD);
    float var = 0.f;
    for (int i = 0; i < DD; ++i) { float z = smean[i] - mu; var += z * z; }
    var *= (1.f / DD);
    sred[0] = mu; sred[1] = var;
  }
  __syncthreads();

  float mu   = sred[0];
  float rstd = rsqrtf(sred[1] + 1e-5f);
  float pln  = (pm - mu) * rstd * ln_g[d] + ln_b[d];

#pragma unroll
  for (int e = 0; e < EE; ++e)
    spart[e][d] = pln * Wg[e * (2 * DD) + d] + pv * Weff[e * DD + d];
  __syncthreads();

  if (d < EE) {
    float le = beff[d];
    for (int i = 0; i < DD; ++i) le += spart[d][i];
    slog[d] = le;
  }
  __syncthreads();

  if (d == 0) {
    float m = slog[0];
#pragma unroll
    for (int e = 1; e < EE; ++e) m = fmaxf(m, slog[e]);
    float ex[EE]; float s = 0.f;
#pragma unroll
    for (int e = 0; e < EE; ++e) { ex[e] = __expf(slog[e] - m); s += ex[e]; }
    float inv = 1.f / s;
#pragma unroll
    for (int e = 0; e < EE; ++e) aff[(size_t)bt * EE + e] = ex[e] * inv;
  }
}

// ---------------- exact top-k per (b,e) via rank; emits weight slot + compact token list ----------------
__global__ __launch_bounds__(1024) void topk_kernel(const float* __restrict__ aff,
                                                    float* __restrict__ selv,
                                                    int* __restrict__ tlist) {
  int be = blockIdx.x;             // b*E + e
  int b = be >> 3, e = be & 7;
  int t = threadIdx.x;
  __shared__ float sa[TT];
  float a = aff[((size_t)b * TT + t) * EE + e];
  sa[t] = a;
  __syncthreads();
  int rank = 0;
  for (int tt = 0; tt < TT; ++tt) {
    float o = sa[tt];
    rank += (o > a) || (o == a && tt < t);
  }
  if (rank < KSEL) {                                   // ranks are a permutation -> exactly KSEL hits
    selv[((size_t)b * TT + t) * EE + e] = a;           // softmax output > 0
    tlist[(be << 8) + rank] = t;
  }
}

// ---------------- expert-major MoE FFN with WMMA ----------------
// One launch per expert (stream-serialized => plain RMW accumulate, no atomics, deterministic).
// Block = 512 threads = 16 waves; 4 patches (M=64). wave: mt = w&3 (patch), wgrp = w>>2 (ntile group).
// The 4 waves sharing an ntile group fetch identical B fragments -> WGP$ hits, ~4x less L2 traffic.
// All hot-loop addresses: one per-lane base pointer per stream + immediate offsets.
#define HRS 392   // LDS H row stride in halves (384 + 8 pad: conflict-free b128 row reads)
#define FRAG 512  // halves per packed fragment (32 lanes x 16)

__global__ __launch_bounds__(512) void ffn_expert_kernel(
    const _Float16* __restrict__ xp, const float* __restrict__ selv,
    const int* __restrict__ tlist,
    const _Float16* __restrict__ wg16, const _Float16* __restrict__ wu16,
    const _Float16* __restrict__ wd16, float* __restrict__ out, int e)
{
  int b  = blockIdx.x >> 6;                 // grid = B * (KSEL/GTOK) = B*64
  int g0 = (blockIdx.x & 63) * GTOK;
  int lane = threadIdx.x & 31;
  int wave = threadIdx.x >> 5;
  int mt   = wave & 3;                      // patch slot
  int wgrp = wave >> 2;                     // 0..3

  __shared__ _Float16 Hs[GTOK][PP * HRS];   // 4 x (16 x 384 padded) f16

  int t = tlist[(((b << 3) | e) << 8) + g0 + mt];       // uniform per wave
  int bt = (b << 10) | t;

  // routing weight for this (token, expert)
  float total = 0.f, mine = 0.f;
#pragma unroll
  for (int ee = 0; ee < EE; ++ee) {
    float s_ = selv[(size_t)bt * EE + ee];
    if (s_ >= 0.f) total += s_;
    if (ee == e) mine = s_;
  }
  float wt = mine * __builtin_amdgcn_rcpf(fmaxf(total, 1e-8f));

  int m0  = (lane >> 4) << 3;
  int col = lane & 15;

  // per-lane base pointers (hot loops use only immediate offsets + one bump per K-step)
  const _Float16* pa = xp + (size_t)bt * PP * DD + (lane & 15) * DD + m0;   // A: row (lane&15), K-phase m0
  const _Float16* pg = wg16 + ((size_t)(e * 144 + wgrp) * FRAG) + lane * 16;
  const _Float16* pu = wu16 + ((size_t)(e * 144 + wgrp) * FRAG) + lane * 16;

  // gate & up: C[16x384] += A[16x192] * B[192x384]   (6 ntiles per wave)
  v8f accg[6], accu[6];
#pragma unroll
  for (int j = 0; j < 6; ++j)
#pragma unroll
    for (int i = 0; i < 8; ++i) { accg[j][i] = 0.f; accu[j][i] = 0.f; }

#pragma unroll 1
  for (int ks = 0; ks < 6; ++ks) {
    v16h a = load_afrag(pa);                // offsets 0 and +32B
#pragma unroll
    for (int j = 0; j < 6; ++j) {
      v16h bg_ = load_frag(pg + j * (4 * FRAG));     // imm offsets j*4096(+16) bytes
      accg[j] = wmma16(a, bg_, accg[j]);
      v16h bu_ = load_frag(pu + j * (4 * FRAG));
      accu[j] = wmma16(a, bu_, accu[j]);
    }
    pa += 32;                               // next K32 slice of the patch row
    pg += 24 * FRAG;                        // next K-step (24 ntiles)
    pu += 24 * FRAG;
  }

  // H = silu(g) * u * wt  -> LDS  (fast v_rcp for the sigmoid)
#pragma unroll
  for (int j = 0; j < 6; ++j) {
    int n = wgrp + 4 * j;
#pragma unroll
    for (int r = 0; r < 8; ++r) {
      float g = accg[j][r], u = accu[j][r];
      float h = g * __builtin_amdgcn_rcpf(1.f + __expf(-g)) * u * wt;
      Hs[mt][(m0 + r) * HRS + n * 16 + col] = (_Float16)h;
    }
  }
  __syncthreads();

  // down: C[16x192] += H[16x384] * B[384x192]   (3 ntiles per wave)
  v8f accd[3];
#pragma unroll
  for (int j = 0; j < 3; ++j)
#pragma unroll
    for (int i = 0; i < 8; ++i) accd[j][i] = 0.f;

  const _Float16* pa2 = &Hs[mt][(lane & 15) * HRS + m0];    // LDS A base (ds imm offsets)
  const _Float16* pd  = wd16 + ((size_t)(e * 144 + wgrp) * FRAG) + lane * 16;

#pragma unroll 1
  for (int ks = 0; ks < 12; ++ks) {
    v16h a2 = load_afrag(pa2);
#pragma unroll
    for (int j = 0; j < 3; ++j) {
      v16h bd = load_frag(pd + j * (4 * FRAG));
      accd[j] = wmma16(a2, bd, accd[j]);
    }
    pa2 += 32;
    pd  += 12 * FRAG;                       // next K-step (12 ntiles)
  }

  // accumulate into out (exclusive per token within this launch; launches ordered by stream)
  int th_ = t >> 5, tw_ = t & (TW_ - 1);
  float* ob = out + (size_t)b * SS * DD;
#pragma unroll
  for (int j = 0; j < 3; ++j) {
    int n = wgrp + 4 * j;
#pragma unroll
    for (int r = 0; r < 8; ++r) {
      int m   = r + m0;
      int row = th_ * 4 + (m >> 2), cc = tw_ * 4 + (m & 3);
      size_t idx = ((size_t)(row * IMGW + cc)) * DD + n * 16 + col;
      ob[idx] += accd[j][r];
    }
  }
}

// ---------------- launcher ----------------
extern "C" void kernel_launch(void* const* d_in, const int* in_sizes, int n_in,
                              void* d_out, int out_size, void* d_ws, size_t ws_size,
                              hipStream_t stream) {
  const float* x     = (const float*)d_in[0];
  const float* ln_g  = (const float*)d_in[1];
  const float* ln_b  = (const float*)d_in[2];
  const float* Wc    = (const float*)d_in[3];
  const float* bc    = (const float*)d_in[4];
  const float* Wg    = (const float*)d_in[5];
  const float* bg    = (const float*)d_in[6];
  const float* Wgate = (const float*)d_in[7];
  const float* Wup   = (const float*)d_in[8];
  const float* Wdown = (const float*)d_in[9];

  char* ws = (char*)d_ws;
  const size_t xp_bytes   = (size_t)BB * TT * PP * DD * 2;   // 25,165,824
  const size_t aff_bytes  = (size_t)BB * TT * EE * 4;        // 131,072
  const size_t selv_bytes = aff_bytes;
  const size_t wpk_halves = (size_t)EE * II * DD;            // 589,824 per tensor
  const size_t wpk_bytes  = wpk_halves * 2;

  _Float16* xp    = (_Float16*)(ws);
  float*    aff   = (float*)(ws + xp_bytes);
  float*    selv  = (float*)(ws + xp_bytes + aff_bytes);
  _Float16* wg16  = (_Float16*)(ws + xp_bytes + aff_bytes + selv_bytes);
  _Float16* wu16  = wg16 + wpk_halves;
  _Float16* wd16  = wu16 + wpk_halves;
  char*     after = ws + xp_bytes + aff_bytes + selv_bytes + 3 * wpk_bytes;
  float*    Weff  = (float*)(after);                          // 8*192*4 = 6144
  float*    beff  = (float*)(after + 8192);                   // 32 B
  int*      tlist = (int*)(after + 8192 + 256);               // B*E*256*4 = 32 KB

  init_selv_kernel<<<(BB * TT * EE + 255) / 256, 256, 0, stream>>>(selv);
  zero_out_kernel<<<(BB * SS * DD / 4) / 256, 256, 0, stream>>>((v4f*)d_out);

  int wtot = (int)wpk_halves;                     // one thread per packed half
  prep_w_kernel<<<(wtot + 255) / 256, 256, 0, stream>>>(Wgate, wg16,  6, 24, II, DD, wtot);
  prep_w_kernel<<<(wtot + 255) / 256, 256, 0, stream>>>(Wup,   wu16,  6, 24, II, DD, wtot);
  prep_w_kernel<<<(wtot + 255) / 256, 256, 0, stream>>>(Wdown, wd16, 12, 12, DD, II, wtot);
  prep_weff_kernel<<<EE, DD, 0, stream>>>(Wc, bc, Wg, bg, Weff, beff);

  stats_gate_kernel<<<BB * TT, DD, 0, stream>>>(x, ln_g, ln_b, Wg, Weff, beff, xp, aff);
  topk_kernel<<<BB * EE, TT, 0, stream>>>(aff, selv, tlist);

  for (int e = 0; e < EE; ++e)
    ffn_expert_kernel<<<BB * (KSEL / GTOK), 512, 0, stream>>>(
        xp, selv, tlist, wg16, wu16, wd16, (float*)d_out, e);
}
